// PerformerAttention_2886218023341
// MI455X (gfx1250) — matrix operand order repaired and checked by reading, not compiled, not run on hardware
//
#include <hip/hip_runtime.h>
#include <cstdint>
#include <cstddef>

// ---------------------------------------------------------------------------
// Types for CDNA5 WMMA bf16 path
// ---------------------------------------------------------------------------
typedef __bf16 bf16;
typedef __attribute__((ext_vector_type(16))) __bf16 bf16x16;
typedef __attribute__((ext_vector_type(8)))  __bf16 bf16x8;
typedef __attribute__((ext_vector_type(8)))  float  f32x8;

union BF16x16 { bf16x16 v; bf16x8 h[2]; };

__device__ __forceinline__ f32x8 wmma_bf16(bf16x16 a, bf16x16 b, f32x8 c) {
  // D = A(16x32 bf16) * B(32x16 bf16) + C(16x16 f32)
  return __builtin_amdgcn_wmma_f32_16x16x32_bf16(
      /*neg_a=*/false, a, /*neg_b=*/false, b,
      /*c_mod=*/(short)0, c, /*reuse_a=*/false, /*reuse_b=*/false);
}

__device__ __forceinline__ float waveSum(float v) {
#pragma unroll
  for (int m = 16; m > 0; m >>= 1) v += __shfl_xor(v, m, 32);
  return v;
}

// Problem constants
#define BATCH 8
#define SEQ   4096
#define EDIM  512
#define MFEAT 256
#define ROWS  (BATCH * SEQ)   // 32768

// ---------------------------------------------------------------------------
// f32 -> bf16 weight conversion
// ---------------------------------------------------------------------------
__global__ void cvt_f32_bf16(const float* __restrict__ src, bf16* __restrict__ dst, int n) {
  int i = blockIdx.x * blockDim.x + threadIdx.x;
  if (i < n) dst[i] = (bf16)src[i];
}

// ---------------------------------------------------------------------------
// LayerNorm: one wave (32 lanes) per 512-elem row, output bf16
// ---------------------------------------------------------------------------
__global__ __launch_bounds__(128) void ln_kernel(
    const float* __restrict__ x, const float* __restrict__ gamma,
    const float* __restrict__ beta, bf16* __restrict__ xn, int rows) {
  int gw = (blockIdx.x * blockDim.x + threadIdx.x) >> 5;
  int lane = threadIdx.x & 31;
  if (gw >= rows) return;
  const float4* xr = (const float4*)(x + (size_t)gw * EDIM);
  float4 d[4];
  float s = 0.f, s2 = 0.f;
#pragma unroll
  for (int i = 0; i < 4; i++) {
    d[i] = xr[lane + 32 * i];
    s  += d[i].x + d[i].y + d[i].z + d[i].w;
    s2 += d[i].x * d[i].x + d[i].y * d[i].y + d[i].z * d[i].z + d[i].w * d[i].w;
  }
  s = waveSum(s); s2 = waveSum(s2);
  float mu  = s  * (1.f / EDIM);
  float var = s2 * (1.f / EDIM) - mu * mu;
  float r = rsqrtf(var + 1e-5f);
  bf16* o = xn + (size_t)gw * EDIM;
#pragma unroll
  for (int i = 0; i < 4; i++) {
    int base = (lane + 32 * i) * 4;
    float vv[4] = {d[i].x, d[i].y, d[i].z, d[i].w};
#pragma unroll
    for (int c = 0; c < 4; c++)
      o[base + c] = (bf16)((vv[c] - mu) * r * gamma[base + c] + beta[base + c]);
  }
}

// ---------------------------------------------------------------------------
// xd[row] = 0.5 * sum(q[row]^2)   (row width 512, bf16 input)
// ---------------------------------------------------------------------------
__global__ __launch_bounds__(128) void xd_kernel(const bf16* __restrict__ q,
                                                 float* __restrict__ xd, int rows) {
  int gw = (blockIdx.x * blockDim.x + threadIdx.x) >> 5;
  int lane = threadIdx.x & 31;
  if (gw >= rows) return;
  const bf16* r = q + (size_t)gw * EDIM;
  float s = 0.f;
#pragma unroll
  for (int i = 0; i < 16; i++) { float v = (float)r[lane + 32 * i]; s += v * v; }
  s = waveSum(s);
  if (lane == 0) xd[gw] = 0.5f * s;
}

// ---------------------------------------------------------------------------
// ksum[b][m] = sum_s kpT[b][m][s]   (row length SEQ, contiguous)
// ---------------------------------------------------------------------------
__global__ __launch_bounds__(128) void ksum_kernel(const bf16* __restrict__ kpT,
                                                   float* __restrict__ ksum, int rows) {
  int gw = (blockIdx.x * blockDim.x + threadIdx.x) >> 5;
  int lane = threadIdx.x & 31;
  if (gw >= rows) return;
  const bf16* r = kpT + (size_t)gw * SEQ;
  float s = 0.f;
  for (int i = 0; i < SEQ / 32; i++) s += (float)r[lane + 32 * i];
  s = waveSum(s);
  if (lane == 0) ksum[gw] = s;
}

// ---------------------------------------------------------------------------
// D[row] = sum_m qp[row][m] * ksum[b][m]   (row width MFEAT)
// ---------------------------------------------------------------------------
__global__ __launch_bounds__(128) void dvec_kernel(const bf16* __restrict__ qp,
                                                   const float* __restrict__ ksum,
                                                   float* __restrict__ D, int rows) {
  int gw = (blockIdx.x * blockDim.x + threadIdx.x) >> 5;
  int lane = threadIdx.x & 31;
  if (gw >= rows) return;
  int b = gw >> 12;  // SEQ == 4096
  const bf16* r = qp + (size_t)gw * MFEAT;
  const float* ks = ksum + (size_t)b * MFEAT;
  float s = 0.f;
#pragma unroll
  for (int i = 0; i < MFEAT / 32; i++) { int idx = lane + 32 * i; s += (float)r[idx] * ks[idx]; }
  s = waveSum(s);
  if (lane == 0) D[gw] = s;
}

// ---------------------------------------------------------------------------
// GEMM with weight-layout B:  C[row,col] = sum_k A[row,k] * W[col,k]
// 128 threads = 4 waves (2x2), block tile 64x64, wave tile 32x32.
// mode 0: outB = acc + bias (also outF f32 if non-null)
// mode 1: outB = exp(acc - xd[row]) * scale      (row-major)
// mode 2: same as 1 but stored transposed: kpT[b][col][s]
// mode 3: outF = resid + acc + bias (f32, final output)
// ---------------------------------------------------------------------------
__global__ __launch_bounds__(128) void gemm_wt(
    const bf16* __restrict__ A, const bf16* __restrict__ W,
    const float* __restrict__ bias, const float* __restrict__ xd,
    const float* __restrict__ resid, bf16* __restrict__ outB,
    float* __restrict__ outF, int N, int K, int mode, float scale) {
  int lane = threadIdx.x & 31;
  int wid  = threadIdx.x >> 5;
  int wm = wid >> 1, wn = wid & 1;
  int m0 = blockIdx.y * 64 + wm * 32;
  int n0 = blockIdx.x * 64 + wn * 32;
  int l15 = lane & 15, lhalf = lane >> 4;
  int ka = lhalf * 16;  // A K-halves: lanes0-15 {0..15}, lanes16-31 {16..31}

  const bf16* Arow0 = A + (size_t)(m0 + l15) * K;
  const bf16* Arow1 = Arow0 + (size_t)16 * K;
  const bf16* Wrow0 = W + (size_t)(n0 + l15) * K + lhalf * 16;  // B: lane=N, K half by lane group
  const bf16* Wrow1 = Wrow0 + (size_t)16 * K;

  f32x8 c[2][2] = {};
  for (int k0 = 0; k0 < K; k0 += 32) {
    __builtin_prefetch(Arow0 + k0 + 128, 0, 0);
    __builtin_prefetch(Wrow0 + k0 + 128, 0, 0);
    BF16x16 a0, a1, b0, b1;
    a0.h[0] = *(const bf16x8*)(Arow0 + k0 + ka);
    a0.h[1] = *(const bf16x8*)(Arow0 + k0 + ka + 8);
    a1.h[0] = *(const bf16x8*)(Arow1 + k0 + ka);
    a1.h[1] = *(const bf16x8*)(Arow1 + k0 + ka + 8);
    b0.v = *(const bf16x16*)(Wrow0 + k0);
    b1.v = *(const bf16x16*)(Wrow1 + k0);
    c[0][0] = wmma_bf16(a0.v, b0.v, c[0][0]);
    c[0][1] = wmma_bf16(a0.v, b1.v, c[0][1]);
    c[1][0] = wmma_bf16(a1.v, b0.v, c[1][0]);
    c[1][1] = wmma_bf16(a1.v, b1.v, c[1][1]);
  }
#pragma unroll
  for (int i = 0; i < 2; i++)
#pragma unroll
    for (int j = 0; j < 2; j++)
#pragma unroll
      for (int v = 0; v < 8; v++) {
        int row = m0 + i * 16 + lhalf * 8 + v;  // C: lanes0-15 M=vgpr, lanes16-31 M=8+vgpr
        int col = n0 + j * 16 + l15;
        float acc = c[i][j][v];
        if (mode == 0) {
          float o = acc + bias[col];
          outB[(size_t)row * N + col] = (bf16)o;
          if (outF) outF[(size_t)row * N + col] = o;
        } else if (mode == 1) {
          outB[(size_t)row * N + col] = (bf16)(__expf(acc - xd[row]) * scale);
        } else if (mode == 2) {
          int b = row >> 12, s = row & 4095;
          outB[((size_t)b * N + col) * SEQ + s] = (bf16)(__expf(acc - xd[row]) * scale);
        } else {
          outF[(size_t)row * N + col] = resid[(size_t)row * N + col] + acc + bias[col];
        }
      }
}

// ---------------------------------------------------------------------------
// Batched GEMM with row-major B:  C[row,n] = sum_k A[row,k] * B[k,n]
// B tile transposed through LDS. 128 threads = 4 waves stacked on M;
// block tile 128M x 32N, wave tile 32x32. Optional per-row 1/(D+eps) scale.
// ---------------------------------------------------------------------------
__global__ __launch_bounds__(128) void gemm_rm(
    const bf16* __restrict__ A, int lda, const bf16* __restrict__ B, int ldb,
    bf16* __restrict__ C, int ldc, const float* __restrict__ Dvec,
    int Mrows, int K, long long strA, long long strB, long long strC) {
  int z = blockIdx.z;
  A += (size_t)z * strA; B += (size_t)z * strB; C += (size_t)z * strC;

  __shared__ bf16 ldsB[32][48];  // [n][k], pitch 48 keeps 32B alignment + dodges conflicts

  int lane = threadIdx.x & 31, wid = threadIdx.x >> 5;
  int l15 = lane & 15, lhalf = lane >> 4;
  int ka = lhalf * 16;
  int m0 = blockIdx.y * 128 + wid * 32;
  int bn = blockIdx.x * 32;
  int tid = threadIdx.x;
  int kr = tid >> 2, nn = (tid & 3) * 8;  // staging: 32 k-rows x 4 chunks of 8 n

  const bf16* Arow0 = A + (size_t)(m0 + l15) * lda;
  const bf16* Arow1 = Arow0 + (size_t)16 * lda;

  f32x8 c[2][2] = {};
  for (int k0 = 0; k0 < K; k0 += 32) {
    // cooperative load of B[k0..k0+31][bn..bn+31], transposed into LDS
    bf16x8 t = *(const bf16x8*)(B + (size_t)(k0 + kr) * ldb + bn + nn);
#pragma unroll
    for (int e = 0; e < 8; e++) ldsB[nn + e][kr] = t[e];
    __syncthreads();

    BF16x16 a0, a1, b0, b1;
    a0.h[0] = *(const bf16x8*)(Arow0 + k0 + ka);
    a0.h[1] = *(const bf16x8*)(Arow0 + k0 + ka + 8);
    a1.h[0] = *(const bf16x8*)(Arow1 + k0 + ka);
    a1.h[1] = *(const bf16x8*)(Arow1 + k0 + ka + 8);
    b0.v = *(const bf16x16*)(&ldsB[l15][lhalf * 16]);
    b1.v = *(const bf16x16*)(&ldsB[16 + l15][lhalf * 16]);
    c[0][0] = wmma_bf16(a0.v, b0.v, c[0][0]);
    c[0][1] = wmma_bf16(a0.v, b1.v, c[0][1]);
    c[1][0] = wmma_bf16(a1.v, b0.v, c[1][0]);
    c[1][1] = wmma_bf16(a1.v, b1.v, c[1][1]);
    __syncthreads();
  }
#pragma unroll
  for (int i = 0; i < 2; i++)
#pragma unroll
    for (int v = 0; v < 8; v++) {
      int row = m0 + i * 16 + lhalf * 8 + v;
      float sc = Dvec ? 1.f / (Dvec[(size_t)z * Mrows + row] + 1e-8f) : 1.f;
#pragma unroll
      for (int j = 0; j < 2; j++) {
        int col = bn + j * 16 + l15;
        C[(size_t)row * ldc + col] = (bf16)(c[i][j][v] * sc);
      }
    }
}

// ---------------------------------------------------------------------------
// Launch
// ---------------------------------------------------------------------------
extern "C" void kernel_launch(void* const* d_in, const int* in_sizes, int n_in,
                              void* d_out, int out_size, void* d_ws, size_t ws_size,
                              hipStream_t stream) {
  const float* x     = (const float*)d_in[0];
  const float* qw    = (const float*)d_in[1];
  const float* qb    = (const float*)d_in[2];
  const float* kw    = (const float*)d_in[3];
  const float* kb    = (const float*)d_in[4];
  const float* vw    = (const float*)d_in[5];
  const float* vb    = (const float*)d_in[6];
  const float* pw    = (const float*)d_in[7];
  const float* pb    = (const float*)d_in[8];
  const float* gamma = (const float*)d_in[9];
  const float* beta  = (const float*)d_in[10];
  const float* w     = (const float*)d_in[11];

  uint8_t* ws = (uint8_t*)d_ws;
  size_t off = 0;
  auto alloc = [&](size_t bytes) { size_t o = off; off = (off + bytes + 255) & ~(size_t)255; return o; };

  bf16* xn    = (bf16*)(ws + alloc((size_t)ROWS * EDIM * 2));
  bf16* qb16  = (bf16*)(ws + alloc((size_t)ROWS * EDIM * 2));
  bf16* kb16  = (bf16*)(ws + alloc((size_t)ROWS * EDIM * 2));
  bf16* vb16  = (bf16*)(ws + alloc((size_t)ROWS * EDIM * 2));
  float* vf32 = (float*)(ws + alloc((size_t)ROWS * EDIM * 4));
  bf16* qp    = (bf16*)(ws + alloc((size_t)ROWS * MFEAT * 2));
  bf16* kpT   = (bf16*)(ws + alloc((size_t)BATCH * MFEAT * SEQ * 2));
  bf16* ynorm = (bf16*)(ws + alloc((size_t)ROWS * EDIM * 2));
  bf16* kptv  = (bf16*)(ws + alloc((size_t)BATCH * MFEAT * EDIM * 2));
  float* xdq  = (float*)(ws + alloc((size_t)ROWS * 4));
  float* xdk  = (float*)(ws + alloc((size_t)ROWS * 4));
  float* Dv   = (float*)(ws + alloc((size_t)ROWS * 4));
  float* ksum = (float*)(ws + alloc((size_t)BATCH * MFEAT * 4));
  bf16* wqB = (bf16*)(ws + alloc((size_t)EDIM * EDIM * 2));
  bf16* wkB = (bf16*)(ws + alloc((size_t)EDIM * EDIM * 2));
  bf16* wvB = (bf16*)(ws + alloc((size_t)EDIM * EDIM * 2));
  bf16* wpB = (bf16*)(ws + alloc((size_t)EDIM * EDIM * 2));
  bf16* wwB = (bf16*)(ws + alloc((size_t)MFEAT * EDIM * 2));

  // 1. weights -> bf16
  int nw = EDIM * EDIM;
  cvt_f32_bf16<<<(nw + 255) / 256, 256, 0, stream>>>(qw, wqB, nw);
  cvt_f32_bf16<<<(nw + 255) / 256, 256, 0, stream>>>(kw, wkB, nw);
  cvt_f32_bf16<<<(nw + 255) / 256, 256, 0, stream>>>(vw, wvB, nw);
  cvt_f32_bf16<<<(nw + 255) / 256, 256, 0, stream>>>(pw, wpB, nw);
  int nww = MFEAT * EDIM;
  cvt_f32_bf16<<<(nww + 255) / 256, 256, 0, stream>>>(w, wwB, nww);

  // 2. LayerNorm
  ln_kernel<<<ROWS / 4, 128, 0, stream>>>(x, gamma, beta, xn, ROWS);

  // 3. q,k,v projections (mode 0); v also in f32 for residual
  dim3 gProj(EDIM / 64, ROWS / 64);
  gemm_wt<<<gProj, 128, 0, stream>>>(xn, wqB, qb, nullptr, nullptr, qb16, nullptr, EDIM, EDIM, 0, 1.f);
  gemm_wt<<<gProj, 128, 0, stream>>>(xn, wkB, kb, nullptr, nullptr, kb16, nullptr, EDIM, EDIM, 0, 1.f);
  gemm_wt<<<gProj, 128, 0, stream>>>(xn, wvB, vb, nullptr, nullptr, vb16, vf32,    EDIM, EDIM, 0, 1.f);

  // 4. xd = 0.5*||.||^2 per row
  xd_kernel<<<ROWS / 4, 128, 0, stream>>>(qb16, xdq, ROWS);
  xd_kernel<<<ROWS / 4, 128, 0, stream>>>(kb16, xdk, ROWS);

  // 5. prm_exp: qp row-major (mode 1), kp transposed -> kpT[b][m][s] (mode 2)
  const float invSqrtM = 0.0625f;  // 1/sqrt(256)
  dim3 gPrm(MFEAT / 64, ROWS / 64);
  gemm_wt<<<gPrm, 128, 0, stream>>>(qb16, wwB, nullptr, xdq, nullptr, qp,  nullptr, MFEAT, EDIM, 1, invSqrtM);
  gemm_wt<<<gPrm, 128, 0, stream>>>(kb16, wwB, nullptr, xdk, nullptr, kpT, nullptr, MFEAT, EDIM, 2, invSqrtM);

  // 6. ksum[b][m], then D[b][s] = qp . ksum
  ksum_kernel<<<(BATCH * MFEAT) / 4, 128, 0, stream>>>(kpT, ksum, BATCH * MFEAT);
  dvec_kernel<<<ROWS / 4, 128, 0, stream>>>(qp, ksum, Dv, ROWS);

  // 7. kptv[b][m][e] = sum_s kpT[b][m][s] * v[b][s][e]
  dim3 gKptv(EDIM / 32, MFEAT / 128, BATCH);
  gemm_rm<<<gKptv, 128, 0, stream>>>(kpT, SEQ, vb16, EDIM, kptv, EDIM, nullptr,
                                     MFEAT, SEQ,
                                     (long long)MFEAT * SEQ, (long long)SEQ * EDIM,
                                     (long long)MFEAT * EDIM);

  // 8. ynorm[b][s][e] = (sum_m qp[b][s][m] * kptv[b][m][e]) / (D + eps)
  dim3 gY(EDIM / 32, SEQ / 128, BATCH);
  gemm_rm<<<gY, 128, 0, stream>>>(qp, MFEAT, kptv, EDIM, ynorm, EDIM, Dv,
                                  SEQ, MFEAT,
                                  (long long)SEQ * MFEAT, (long long)MFEAT * EDIM,
                                  (long long)SEQ * EDIM);

  // 9. out = v + ynorm @ pw^T + pb  (mode 3, f32 output)
  gemm_wt<<<gProj, 128, 0, stream>>>(ynorm, wpB, pb, nullptr, vf32, nullptr,
                                     (float*)d_out, EDIM, EDIM, 3, 1.f);

  (void)in_sizes; (void)n_in; (void)out_size; (void)ws_size;
}
